// PointNetSetAbstraction_65171833749934
// MI455X (gfx1250) — matrix operand order repaired and checked by reading, not compiled, not run on hardware
//
#include <hip/hip_runtime.h>

typedef _Float16 half_t;
typedef __attribute__((ext_vector_type(16))) _Float16 v16h;
typedef __attribute__((ext_vector_type(8)))  _Float16 v8h;
typedef __attribute__((ext_vector_type(8)))  float    v8f;

#define NB      8
#define NPTS    4096
#define NPOINT  1024
#define NSAMPLE 32
#define MROWS   (NB * NPOINT * NSAMPLE)   /* 262144 */
#define RAD2    0.04f
#define BN_EPS  1e-5f

/* ------------------------------------------------------------------ */
/* 1) Farthest point sampling: one block per batch, xyz staged in LDS. */
/*    Serial 1024-step loop; LDS tree argmax with first-index ties.    */
/* ------------------------------------------------------------------ */
__global__ void fps_kernel(const float* __restrict__ xyz,
                           float* __restrict__ newxyz_ws,
                           float* __restrict__ out_newxyz) {
  const int b = blockIdx.x;
  const int tid = threadIdx.x;           // 256 threads
  __shared__ float px[NPTS], py[NPTS], pz[NPTS];
  __shared__ float rv[256];
  __shared__ int   ri[256];
  __shared__ int   sfid[NPOINT];
  __shared__ int   curFar;

  const float* Xb = xyz + (size_t)b * 3 * NPTS;
  for (int n = tid; n < NPTS; n += 256) {
    px[n] = Xb[n];
    py[n] = Xb[NPTS + n];
    pz[n] = Xb[2 * NPTS + n];
  }
  float dist[16];
#pragma unroll
  for (int i = 0; i < 16; ++i) dist[i] = 1e10f;
  if (tid == 0) curFar = 0;
  __syncthreads();

  for (int s = 0; s < NPOINT; ++s) {
    const int far = curFar;
    if (tid == 0) sfid[s] = far;
    const float cx = px[far], cy = py[far], cz = pz[far];
    float bv = -1.0f;
    int   bi = 0;
#pragma unroll
    for (int i = 0; i < 16; ++i) {
      const int n = tid + (i << 8);
      const float dx = px[n] - cx, dy = py[n] - cy, dz = pz[n] - cz;
      const float d  = dx * dx + dy * dy + dz * dz;
      const float nd = fminf(dist[i], d);
      dist[i] = nd;
      if (nd > bv) { bv = nd; bi = n; }
    }
    rv[tid] = bv; ri[tid] = bi;
    __syncthreads();
    for (int off = 128; off > 0; off >>= 1) {
      if (tid < off) {
        const float ov = rv[tid + off];
        const int   oi = ri[tid + off];
        if (ov > rv[tid] || (ov == rv[tid] && oi < ri[tid])) { rv[tid] = ov; ri[tid] = oi; }
      }
      __syncthreads();
    }
    if (tid == 0) curFar = ri[0];
    __syncthreads();
  }

  // Emit new_xyz: row-major (b,s,3) scratch for ball-query + transposed (b,3,s) output.
  for (int s = tid; s < NPOINT; s += 256) {
    const int n = sfid[s];
    const float x = px[n], y = py[n], z = pz[n];
    float* w = newxyz_ws + ((size_t)b * NPOINT + s) * 3;
    w[0] = x; w[1] = y; w[2] = z;
    out_newxyz[(size_t)b * 3 * NPOINT + 0 * NPOINT + s] = x;
    out_newxyz[(size_t)b * 3 * NPOINT + 1 * NPOINT + s] = y;
    out_newxyz[(size_t)b * 3 * NPOINT + 2 * NPOINT + s] = z;
  }
}

/* ------------------------------------------------------------------ */
/* 2) Ball query: one wave32 per query; ballot + prefix popcount       */
/*    reproduces "first NSAMPLE in index order, pad with first".       */
/* ------------------------------------------------------------------ */
__global__ void ballquery_kernel(const float* __restrict__ xyz,
                                 const float* __restrict__ newxyz,
                                 int* __restrict__ idx) {
  const int lane = threadIdx.x & 31;
  const int q = blockIdx.x * 8 + (threadIdx.x >> 5);   // 0..8191
  const int b = q >> 10;
  const float* c = newxyz + (size_t)q * 3;
  const float cx = c[0], cy = c[1], cz = c[2];
  const float* Xb = xyz + (size_t)b * 3 * NPTS;

  int cnt = 0;
  int first = NPTS - 1;
  for (int base = 0; base < NPTS && cnt < NSAMPLE; base += 32) {
    const int n = base + lane;
    const float dx = Xb[n] - cx, dy = Xb[NPTS + n] - cy, dz = Xb[2 * NPTS + n] - cz;
    const bool in = (dx * dx + dy * dy + dz * dz) <= RAD2;
    const unsigned mask = (unsigned)__ballot(in);   // wave32: low 32 bits
    if (in) {
      const int pos = cnt + __popc(mask & ((1u << lane) - 1u));
      if (pos < NSAMPLE) idx[(size_t)q * NSAMPLE + pos] = n;
    }
    if (cnt == 0 && mask != 0u) first = base + __builtin_ctz(mask);
    cnt += __popc(mask);
  }
  if (cnt > NSAMPLE) cnt = NSAMPLE;
  if (lane < NSAMPLE - cnt) idx[(size_t)q * NSAMPLE + cnt + lane] = first;
}

/* ------------------------------------------------------------------ */
/* 3) Weight pre-conversion: f32 -> f16 once per launch.               */
/*    w1: 64x64 = 4096, w2: 128x64 = 8192 elements.                    */
/* ------------------------------------------------------------------ */
__global__ void convw_kernel(const float* __restrict__ w1,
                             const float* __restrict__ w2,
                             half_t* __restrict__ w1h,
                             half_t* __restrict__ w2h) {
  const int i = blockIdx.x * 256 + threadIdx.x;   // < 12288
  if (i < 4096) {
    w1h[i] = (half_t)w1[i];
  } else {
    w2h[i - 4096] = (half_t)w2[i - 4096];
  }
}

/* ------------------------------------------------------------------ */
/* 4) Layer 0: gather + 6->64 linear (K=6 too skinny for WMMA).        */
/* ------------------------------------------------------------------ */
__global__ void layer0_kernel(const float* __restrict__ xyz,
                              const float* __restrict__ points,
                              const float* __restrict__ newxyz,
                              const int* __restrict__ idx,
                              const float* __restrict__ w0,
                              const float* __restrict__ b0,
                              half_t* __restrict__ z0) {
  const int gid = blockIdx.x * 256 + threadIdx.x;   // < MROWS
  const int b = gid >> 15;
  const int q = gid >> 5;
  int n = idx[gid];
  if (n > NPTS - 1) n = NPTS - 1;
  const float* Xb = xyz + (size_t)b * 3 * NPTS;
  const float* Pb = points + (size_t)b * 3 * NPTS;
  const float* c  = newxyz + (size_t)q * 3;
  float f[6];
  f[0] = Xb[n] - c[0];
  f[1] = Xb[NPTS + n] - c[1];
  f[2] = Xb[2 * NPTS + n] - c[2];
  f[3] = Pb[n];
  f[4] = Pb[NPTS + n];
  f[5] = Pb[2 * NPTS + n];
  half_t* zr = z0 + (size_t)gid * 64;
  for (int oc = 0; oc < 64; ++oc) {
    float acc = b0[oc];
#pragma unroll
    for (int j = 0; j < 6; ++j) acc += f[j] * w0[oc * 6 + j];
    zr[oc] = (half_t)acc;
  }
}

/* ------------------------------------------------------------------ */
/* 5) BN statistics: deterministic two stage (block partials, then     */
/*    single-block reduce). No float atomics -> replay-stable.         */
/* ------------------------------------------------------------------ */
template <int C>
__global__ void stats_kernel(const half_t* __restrict__ Z, float* __restrict__ part) {
  constexpr int G = 256 / C;
  const int tid = threadIdx.x;
  const int c = tid % C;
  const int g = tid / C;
  const size_t rowBase = (size_t)blockIdx.x * 1024;
  float s1 = 0.f, s2 = 0.f;
  for (int r = g; r < 1024; r += G) {
    const float v = (float)Z[(rowBase + r) * C + c];
    s1 += v; s2 += v * v;
  }
  __shared__ float sh[256];
  sh[tid] = s1; __syncthreads();
  if (tid < C) {
#pragma unroll
    for (int gg = 1; gg < G; ++gg) s1 += sh[gg * C + tid];
    part[(size_t)blockIdx.x * 2 * C + tid] = s1;
  }
  __syncthreads();
  sh[tid] = s2; __syncthreads();
  if (tid < C) {
#pragma unroll
    for (int gg = 1; gg < G; ++gg) s2 += sh[gg * C + tid];
    part[(size_t)blockIdx.x * 2 * C + C + tid] = s2;
  }
}

template <int C>
__global__ void reduce_stats_kernel(const float* __restrict__ part, float* __restrict__ sums) {
  const int t = threadIdx.x;   // 0 .. 2C-1
  float s = 0.f;
  for (int bk = 0; bk < 256; ++bk) s += part[(size_t)bk * 2 * C + t];
  sums[t] = s;
}

/* ------------------------------------------------------------------ */
/* 6) BN + ReLU applied in place on the f16 activation buffer.         */
/* ------------------------------------------------------------------ */
template <int C>
__global__ void normrelu_kernel(half_t* __restrict__ Z,
                                const float* __restrict__ sums,
                                const float* __restrict__ gamma,
                                const float* __restrict__ beta) {
  const size_t i = (size_t)blockIdx.x * 256 + threadIdx.x;
  const int c = (int)(i % C);
  const float inv  = 1.0f / (float)MROWS;
  const float mean = sums[c] * inv;
  const float var  = sums[C + c] * inv - mean * mean;
  const float sc   = gamma[c] * rsqrtf(var + BN_EPS);
  float v = (float)Z[i];
  v = (v - mean) * sc + beta[c];
  Z[i] = (half_t)fmaxf(v, 0.0f);
}

/* ------------------------------------------------------------------ */
/* 7) WMMA GEMM: Z = relu'd X (M x 64, f16) * W^T (COUT x 64, f16) + b */
/*    Wave computes a 16 x COUT strip via v_wmma_f32_16x16x32_f16.     */
/*    A frag: lanes 0-15 row r K 0-15; lanes 16-31 row r K 16-31 */
/*    B frag: lane r -> col, K 0-15; lane r+16 -> col, K 16-31.        */
/*    C/D   : lane t col t%16, VGPR v row v + 8*(t>>4).                */
/*    Weights pre-converted to f16: B fragment is one aligned v16h.    */
/* ------------------------------------------------------------------ */
template <int COUT>
__global__ void gemm_kernel(const half_t* __restrict__ X,
                            const half_t* __restrict__ W16,
                            const float* __restrict__ bias,
                            half_t* __restrict__ Z) {
  constexpr int NT = COUT / 16;
  const int lane = threadIdx.x & 31;
  const int wave = threadIdx.x >> 5;
  const int r  = lane & 15;
  const int hi = lane >> 4;
  const int rowBase = (blockIdx.x * 8 + wave) * 16;

  union AF { v16h v; v8h h[2]; };
  AF a[2];
#pragma unroll
  for (int kb = 0; kb < 2; ++kb) {
    const half_t* ap = X + (size_t)(rowBase + r) * 64 + kb * 32 + hi * 16;
    a[kb].h[0] = *(const v8h*)(ap);
    a[kb].h[1] = *(const v8h*)(ap + 8);
  }

  v16h bfr[NT][2];
#pragma unroll
  for (int nt = 0; nt < NT; ++nt) {
    const int n = nt * 16 + r;
#pragma unroll
    for (int kb = 0; kb < 2; ++kb) {
      // 16 contiguous halves, 32B aligned: K = kb*32 + hi*16 + [0..15]
      bfr[nt][kb] = *(const v16h*)(W16 + (size_t)n * 64 + kb * 32 + hi * 16);
    }
  }

  v8f acc[NT];
#pragma unroll
  for (int nt = 0; nt < NT; ++nt) {
    const float bb = bias[nt * 16 + r];
#pragma unroll
    for (int v = 0; v < 8; ++v) acc[nt][v] = bb;
  }

#pragma unroll
  for (int kb = 0; kb < 2; ++kb) {
#pragma unroll
    for (int nt = 0; nt < NT; ++nt) {
      acc[nt] = __builtin_amdgcn_wmma_f32_16x16x32_f16(
          /*neg_a=*/false, a[kb].v, /*neg_b=*/false, bfr[nt][kb],
          /*c_mod=*/(short)0, acc[nt], /*reuse_a=*/false, /*reuse_b=*/false);
    }
  }

#pragma unroll
  for (int nt = 0; nt < NT; ++nt) {
#pragma unroll
    for (int v = 0; v < 8; ++v) {
      const int m = rowBase + v + hi * 8;
      Z[(size_t)m * COUT + nt * 16 + r] = (half_t)acc[nt][v];
    }
  }
}

/* ------------------------------------------------------------------ */
/* 8) Final: BN+ReLU on z2, max over K=32, transposed (b, oc, s) out.  */
/* ------------------------------------------------------------------ */
__global__ void final_kernel(const half_t* __restrict__ Z2,
                             const float* __restrict__ sums2,
                             const float* __restrict__ gamma,
                             const float* __restrict__ beta,
                             float* __restrict__ out) {
  const int gid = blockIdx.x * 256 + threadIdx.x;   // < 8*1024*128
  const int oc = gid & 127;
  const int q  = gid >> 7;      // 0..8191
  const int b  = q >> 10;
  const int s  = q & 1023;
  const float inv  = 1.0f / (float)MROWS;
  const float mean = sums2[oc] * inv;
  const float var  = sums2[128 + oc] * inv - mean * mean;
  const float sc   = gamma[oc] * rsqrtf(var + BN_EPS);
  const float sh   = beta[oc];
  float m = 0.0f;   // max over relu'd values is >= 0
  const half_t* zp = Z2 + (size_t)q * NSAMPLE * 128 + oc;
#pragma unroll 4
  for (int k = 0; k < NSAMPLE; ++k) {
    float v = (float)zp[(size_t)k * 128];
    v = (v - mean) * sc + sh;
    m = fmaxf(m, v);
  }
  out[24576 + (size_t)b * 128 * 1024 + (size_t)oc * 1024 + s] = m;
}

/* ------------------------------------------------------------------ */
/* Host side. Workspace layout (bytes):                                */
/*   0        newxyz   f32 [8*1024*3]      (98,304)                    */
/*   98304    idx      i32 [262144]        (1,048,576)                 */
/*   1146880  sums     f32 [512]           (2,048)                     */
/*   1148928  partials f32 [256*256]       (262,144)                   */
/*   1411072  w1h      f16 [4096]          (8,192)                     */
/*   1419264  w2h      f16 [8192]          (16,384)                    */
/*   1435648  z0       f16 [M*64]          (33,554,432)                */
/*   34990080 z1       f16 [M*64]          (33,554,432)                */
/*   68544512 z2       f16 [M*128]         (67,108,864)  end ~135.7 MB */
/* ------------------------------------------------------------------ */
extern "C" void kernel_launch(void* const* d_in, const int* in_sizes, int n_in,
                              void* d_out, int out_size, void* d_ws, size_t ws_size,
                              hipStream_t stream) {
  (void)in_sizes; (void)n_in; (void)out_size; (void)ws_size;

  const float* xyz    = (const float*)d_in[0];
  const float* points = (const float*)d_in[1];
  const float* w0  = (const float*)d_in[2];
  const float* b0  = (const float*)d_in[3];
  const float* g0  = (const float*)d_in[4];
  const float* be0 = (const float*)d_in[5];
  const float* w1  = (const float*)d_in[6];
  const float* b1  = (const float*)d_in[7];
  const float* g1  = (const float*)d_in[8];
  const float* be1 = (const float*)d_in[9];
  const float* w2  = (const float*)d_in[10];
  const float* b2  = (const float*)d_in[11];
  const float* g2  = (const float*)d_in[12];
  const float* be2 = (const float*)d_in[13];

  char* ws = (char*)d_ws;
  float*  newxyz = (float*)ws;
  int*    idx    = (int*)(ws + 98304);
  float*  sums   = (float*)(ws + 1146880);   // 3 layers: [0..128) [128..256) [256..512)
  float*  part   = (float*)(ws + 1148928);
  half_t* w1h    = (half_t*)(ws + 1411072);
  half_t* w2h    = (half_t*)(ws + 1419264);
  half_t* z0     = (half_t*)(ws + 1435648);
  half_t* z1     = (half_t*)(ws + 34990080);
  half_t* z2     = (half_t*)(ws + 68544512);
  float*  out    = (float*)d_out;

  convw_kernel<<<48, 256, 0, stream>>>(w1, w2, w1h, w2h);
  fps_kernel<<<NB, 256, 0, stream>>>(xyz, newxyz, out);
  ballquery_kernel<<<(NB * NPOINT) / 8, 256, 0, stream>>>(xyz, newxyz, idx);
  layer0_kernel<<<MROWS / 256, 256, 0, stream>>>(xyz, points, newxyz, idx, w0, b0, z0);

  stats_kernel<64><<<256, 256, 0, stream>>>(z0, part);
  reduce_stats_kernel<64><<<1, 128, 0, stream>>>(part, sums);
  normrelu_kernel<64><<<(MROWS * 64) / 256, 256, 0, stream>>>(z0, sums, g0, be0);

  gemm_kernel<64><<<MROWS / 128, 256, 0, stream>>>(z0, w1h, b1, z1);
  stats_kernel<64><<<256, 256, 0, stream>>>(z1, part);
  reduce_stats_kernel<64><<<1, 128, 0, stream>>>(part, sums + 128);
  normrelu_kernel<64><<<(MROWS * 64) / 256, 256, 0, stream>>>(z1, sums + 128, g1, be1);

  gemm_kernel<128><<<MROWS / 128, 256, 0, stream>>>(z1, w2h, b2, z2);
  stats_kernel<128><<<256, 256, 0, stream>>>(z2, part);
  reduce_stats_kernel<128><<<1, 256, 0, stream>>>(part, sums + 256);

  final_kernel<<<(NB * NPOINT * 128) / 256, 256, 0, stream>>>(z2, sums + 256, g2, be2, out);
}
